// RecurrentDecoderLSTM_3736621547810
// MI455X (gfx1250) — matrix operand order repaired and checked by reading, not compiled, not run on hardware
//
#include <hip/hip_runtime.h>
#include <stddef.h>

// Problem constants (fixed by the reference).
constexpr int CB = 512;   // batch
constexpr int CL = 512;   // hidden size L
constexpr int CF = 128;   // feature size F
constexpr int CT = 512;   // timesteps
constexpr int ROWS = 16;  // batch rows per workgroup (one WMMA M tile)

constexpr int HLD = CL + 8;   // bf16 h row stride (pad to stagger banks, keep 16B align)
constexpr int XLD = CF + 8;   // bf16 x row stride
constexpr int CLD = CL + 4;   // f32 c row stride

typedef __attribute__((ext_vector_type(16))) __bf16 v16bf;
typedef __attribute__((ext_vector_type(8)))  __bf16 v8bf;
typedef __attribute__((ext_vector_type(8)))  float  v8f;

static __device__ __forceinline__ v8f vzero() {
  return (v8f){0.f, 0.f, 0.f, 0.f, 0.f, 0.f, 0.f, 0.f};
}

static __device__ __forceinline__ float sigm(float x) {
  return 1.0f / (1.0f + __expf(-x));
}
static __device__ __forceinline__ float tanh_fast(float x) {
  return 2.0f / (1.0f + __expf(-2.0f * x)) - 1.0f;
}

// Accumulate a 16x16 f32 tile: A = 16 x kdim (bf16, LDS row base already offset
// by lane-half: arow = &As[lane%16][ (lane/16)*8 ]), B^T = weight rows W[n][k]
// (bf16, global; wrow = &W[(n0 + lane%16)*kdim + (lane/16)*16]).
// A-fragment per ISA 7.12.2: lane<16 holds K {kb..kb+7, kb+16..kb+23},
// lane>=16 holds K {kb+8..kb+15, kb+24..kb+31}.
// B-fragment: same K order as A: lane<16 holds K {kb..kb+7, kb+16..kb+23},
// lane>=16 holds K {kb+8..kb+15, kb+24..kb+31} of the K-major weight row.
static __device__ __forceinline__ v8f wmma_k(v8f acc,
                                             const __bf16* __restrict__ arow,
                                             const __bf16* __restrict__ wrow,
                                             int kdim) {
#pragma unroll 4
  for (int kb = 0; kb < kdim; kb += 32) {
    v8bf alo = *(const v8bf*)(arow + kb);
    v8bf ahi = *(const v8bf*)(arow + kb + 16);
    v16bf A = __builtin_shufflevector(alo, ahi, 0, 1, 2, 3, 4, 5, 6, 7,
                                      8, 9, 10, 11, 12, 13, 14, 15);
    const __bf16* wb = wrow - 8 * ((threadIdx.x >> 4) & 1) + kb;
    v8bf blo = *(const v8bf*)(wb);
    v8bf bhi = *(const v8bf*)(wb + 16);
    v16bf Bf = __builtin_shufflevector(blo, bhi, 0, 1, 2, 3, 4, 5, 6, 7,
                                       8, 9, 10, 11, 12, 13, 14, 15);
    acc = __builtin_amdgcn_wmma_f32_16x16x32_bf16(
        /*neg_a=*/false, A, /*neg_b=*/false, Bf,
        /*c_mod=*/(short)0, acc, /*reuse_a=*/false, /*reuse_b=*/false);
  }
  return acc;
}

// Dense layer: x = h @ W_dense^T + b_dense.  Waves 0..7 each own one 16-col
// tile of the F=128 output.  Writes bf16 x to LDS; if t >= 0 also writes f32
// to out[b, t, f].
static __device__ __forceinline__ void dense_step(
    int wave, int lrow, int lsel,
    const __bf16 (*hS)[HLD], __bf16 (*xS)[XLD],
    const __bf16* __restrict__ Wd, const float* __restrict__ b_dense,
    float* __restrict__ out, int b0, int t) {
  if (wave >= 8) return;
  const int ncol = wave * 16;
  v8f acc = vzero();
  acc = wmma_k(acc, &hS[lrow][lsel * 8],
               Wd + (size_t)(ncol + lrow) * CL + lsel * 16, CL);
  const float bd = b_dense[ncol + lrow];
#pragma unroll
  for (int r = 0; r < 8; ++r) {
    const int M = r + 8 * lsel;  // C/D layout: VGPR r, lane half -> row
    const float xv = acc[r] + bd;
    xS[M][ncol + lrow] = (__bf16)xv;
    if (t >= 0) {
      out[(size_t)(b0 + M) * ((size_t)CT * CF) + (size_t)t * CF +
          (ncol + lrow)] = xv;
    }
  }
}

__global__ void __launch_bounds__(512)
lstm_decoder_kernel(const float* __restrict__ h0, const float* __restrict__ c0,
                    const __bf16* __restrict__ Wih,
                    const __bf16* __restrict__ Whh,
                    const __bf16* __restrict__ Wd,
                    const float* __restrict__ b_ih,
                    const float* __restrict__ b_hh,
                    const float* __restrict__ b_dense,
                    float* __restrict__ out) {
  __shared__ __align__(16) __bf16 hS[ROWS][HLD];
  __shared__ __align__(16) __bf16 xS[ROWS][XLD];
  __shared__ __align__(16) float  cS[ROWS][CLD];

  const int b0 = blockIdx.x * ROWS;
  const int tid = threadIdx.x;
  const int wave = tid >> 5;
  const int lane = tid & 31;
  const int lrow = lane & 15;  // M (A/C) or N (B/C) index within a 16-tile
  const int lsel = lane >> 4;  // lane half

  // ---- Load initial state: h0 -> bf16 LDS, c0 -> f32 LDS ----
  for (int i = tid; i < ROWS * CL; i += 512) {
    const int r = i >> 9;       // / CL
    const int cix = i & (CL - 1);
    hS[r][cix] = (__bf16)h0[(size_t)(b0 + r) * CL + cix];
    cS[r][cix] = c0[(size_t)(b0 + r) * CL + cix];
  }
  __syncthreads();

  // Initial x = dense(h0)  (not emitted to output)
  dense_step(wave, lrow, lsel, hS, xS, Wd, b_dense, out, b0, /*t=*/-1);

  const __bf16* const ax = &xS[lrow][lsel * 8];
  const __bf16* const ah = &hS[lrow][lsel * 8];

  for (int t = 0; t < CT; ++t) {
    __syncthreads();  // new x / h visible to everyone

    // ---- gates = x @ Wih^T + h @ Whh^T  (f32 accumulate) ----
    v8f gI[2], gF[2], gG[2], gO[2];
#pragma unroll
    for (int jj = 0; jj < 2; ++jj) {
      const int nrow = (wave * 2 + jj) * 16 + lrow;  // per-lane B row (gate N)
      v8f a0 = vzero(), a1 = vzero(), a2 = vzero(), a3 = vzero();
      a0 = wmma_k(a0, ax, Wih + (size_t)(0 * CL + nrow) * CF + lsel * 16, CF);
      a1 = wmma_k(a1, ax, Wih + (size_t)(1 * CL + nrow) * CF + lsel * 16, CF);
      a2 = wmma_k(a2, ax, Wih + (size_t)(2 * CL + nrow) * CF + lsel * 16, CF);
      a3 = wmma_k(a3, ax, Wih + (size_t)(3 * CL + nrow) * CF + lsel * 16, CF);
      a0 = wmma_k(a0, ah, Whh + (size_t)(0 * CL + nrow) * CL + lsel * 16, CL);
      a1 = wmma_k(a1, ah, Whh + (size_t)(1 * CL + nrow) * CL + lsel * 16, CL);
      a2 = wmma_k(a2, ah, Whh + (size_t)(2 * CL + nrow) * CL + lsel * 16, CL);
      a3 = wmma_k(a3, ah, Whh + (size_t)(3 * CL + nrow) * CL + lsel * 16, CL);
      gI[jj] = a0; gF[jj] = a1; gG[jj] = a2; gO[jj] = a3;
    }

    __syncthreads();  // all reads of old h complete before h is overwritten

    // ---- LSTM cell (f32), update c (f32 LDS) and h (bf16 LDS) ----
#pragma unroll
    for (int jj = 0; jj < 2; ++jj) {
      const int n = (wave * 2 + jj) * 16 + lrow;  // element N index
      const float bI = b_ih[0 * CL + n] + b_hh[0 * CL + n];
      const float bF = b_ih[1 * CL + n] + b_hh[1 * CL + n];
      const float bG = b_ih[2 * CL + n] + b_hh[2 * CL + n];
      const float bO = b_ih[3 * CL + n] + b_hh[3 * CL + n];
#pragma unroll
      for (int r = 0; r < 8; ++r) {
        const int M = r + 8 * lsel;
        const float iv = sigm(gI[jj][r] + bI);
        const float fv = sigm(gF[jj][r] + bF);
        const float gv = tanh_fast(gG[jj][r] + bG);
        const float ov = sigm(gO[jj][r] + bO);
        const float c2 = fv * cS[M][n] + iv * gv;
        cS[M][n] = c2;
        hS[M][n] = (__bf16)(ov * tanh_fast(c2));
      }
    }

    __syncthreads();  // new h visible

    // ---- x = dense(h), emit out[:, t, :] ----
    dense_step(wave, lrow, lsel, hS, xS, Wd, b_dense, out, b0, t);
  }
}

// One-shot fp32 -> bf16 weight conversion into workspace (weights then live in
// the 192MB L2 for the whole recurrence).
__global__ void convert_weights_kernel(const float* __restrict__ Wih,
                                       const float* __restrict__ Whh,
                                       const float* __restrict__ Wd,
                                       __bf16* __restrict__ oih,
                                       __bf16* __restrict__ ohh,
                                       __bf16* __restrict__ od) {
  const int i = blockIdx.x * blockDim.x + threadIdx.x;
  const int NIH = 4 * CL * CF;
  const int NHH = 4 * CL * CL;
  const int ND = CF * CL;
  if (i < NIH) oih[i] = (__bf16)Wih[i];
  if (i < NHH) ohh[i] = (__bf16)Whh[i];
  if (i < ND)  od[i]  = (__bf16)Wd[i];
}

extern "C" void kernel_launch(void* const* d_in, const int* in_sizes, int n_in,
                              void* d_out, int out_size, void* d_ws,
                              size_t ws_size, hipStream_t stream) {
  const float* h0      = (const float*)d_in[0];
  const float* c0      = (const float*)d_in[1];
  const float* W_ih    = (const float*)d_in[2];
  const float* W_hh    = (const float*)d_in[3];
  const float* b_ih    = (const float*)d_in[4];
  const float* b_hh    = (const float*)d_in[5];
  const float* W_dense = (const float*)d_in[6];
  const float* b_dense = (const float*)d_in[7];
  float* out = (float*)d_out;

  __bf16* wih_b = (__bf16*)d_ws;                       // 4L*F  = 512 KB
  __bf16* whh_b = wih_b + (size_t)4 * CL * CF;         // 4L*L  = 2 MB
  __bf16* wd_b  = whh_b + (size_t)4 * CL * CL;         // F*L   = 128 KB

  const int nmax = 4 * CL * CL;
  convert_weights_kernel<<<(nmax + 255) / 256, 256, 0, stream>>>(
      W_ih, W_hh, W_dense, wih_b, whh_b, wd_b);

  lstm_decoder_kernel<<<CB / ROWS, 512, 0, stream>>>(
      h0, c0, wih_b, whh_b, wd_b, b_ih, b_hh, b_dense, out);
}
